// DeepFFM_58574763983367
// MI455X (gfx1250) — matrix operand order repaired and checked by reading, not compiled, not run on hardware
//
#include <hip/hip_runtime.h>
#include <hip/hip_bf16.h>
#include <stdint.h>

// ---------------------------------------------------------------------------
// DeepFFM on gfx1250 (MI455X): bf16 WMMA GEMM pipeline, f32 accumulation,
// Tensor Data Mover (TDM) double-buffered LDS staging.
// ---------------------------------------------------------------------------

typedef __attribute__((ext_vector_type(16))) __bf16        v16bf;
typedef __attribute__((ext_vector_type(8)))  float         v8f;
typedef __attribute__((ext_vector_type(4)))  unsigned int  u32x4;
typedef __attribute__((ext_vector_type(8)))  int           i32x8;
typedef __attribute__((ext_vector_type(4)))  int           i32x4;

#define BATCH   4096
#define FSZ     1000
#define KPAD    1024
#define NFIELDS 39
#define KFF     40
#define L0      1024
#define L1      512
#define L2      256

__device__ __forceinline__ __bf16 f2bf(float x) {
    union { float f; uint32_t u; } a; a.f = x;
    uint32_t r = a.u + 0x7FFFu + ((a.u >> 16) & 1u);   // round-to-nearest-even
    union { uint16_t s; __bf16 b; } o; o.s = (uint16_t)(r >> 16);
    return o.b;
}
__device__ __forceinline__ float bf2f(__bf16 x) {
    union { uint16_t s; __bf16 b; } i; i.b = x;
    union { float f; uint32_t u; } o; o.u = ((uint32_t)i.s) << 16;
    return o.f;
}

// --------------------------- TDM tile load ---------------------------------
// DMA a [128 x 32] bf16 tile (row stride = lda elements) from global memory
// into LDS at byte offset lds_addr. D# per CDNA5 ISA ch.8 (2D, groups 2/3 = 0).
__device__ __forceinline__ void tdm_load_tile(const __bf16* gbase, size_t elem_off,
                                              int lda_elems, uint32_t lds_addr) {
    uint64_t gaddr = (uint64_t)(uintptr_t)gbase + elem_off * 2u;

    u32x4 g0;
    g0[0] = 1u;                                            // count=1 (user D#)
    g0[1] = lds_addr;                                      // lds_addr [63:32]
    g0[2] = (uint32_t)gaddr;                               // global_addr[31:0]
    g0[3] = (uint32_t)((gaddr >> 32) & 0x01FFFFFFu)        // global_addr[56:32]
          | (2u << 30);                                    // type=2 ("image")

    uint32_t d0 = (uint32_t)lda_elems;                     // tensor_dim0 (elems)
    uint32_t d1 = 128u;                                    // tensor_dim1 (rows)
    uint64_t s0 = (uint64_t)(uint32_t)lda_elems;           // tensor_dim0_stride
    i32x8 g1;
    g1[0] = (int)(1u << 16);                               // data_size=1 (2B)
    g1[1] = (int)((d0 & 0xFFFFu) << 16);                   // tensor_dim0[15:0]
    g1[2] = (int)(((d0 >> 16) & 0xFFFFu) | ((d1 & 0xFFFFu) << 16));
    g1[3] = (int)(((d1 >> 16) & 0xFFFFu) | (32u << 16));   // tile_dim0 = 32
    g1[4] = (int)128u;                                     // tile_dim1 = 128
    g1[5] = (int)(uint32_t)(s0 & 0xFFFFFFFFu);             // stride0[31:0]
    g1[6] = (int)(uint32_t)((s0 >> 32) & 0xFFFFu);         // stride0[47:32]
    g1[7] = 0;                                             // dim1_stride = 0 (2D)

    i32x4 z4 = {0, 0, 0, 0};
    i32x8 z8 = {0, 0, 0, 0, 0, 0, 0, 0};
    // 6-arg variant (clang-23 / therock-10 headers): (g0, g1, g2, g3, g4, cpol)
    __builtin_amdgcn_tensor_load_to_lds(g0, g1, z4, z4, z8, 0);
}

// --------------------------- conversion kernels ----------------------------

// X (BATCH x FSZ f32) -> Xb (BATCH x KPAD bf16), zero-padded K tail.
__global__ __launch_bounds__(256) void k_pad_x(const float* __restrict__ X,
                                               __bf16* __restrict__ Xb) {
    int idx = blockIdx.x * 256 + threadIdx.x;              // over BATCH*KPAD
    int r = idx >> 10, c = idx & 1023;
    float v = (c < FSZ) ? X[(size_t)r * FSZ + c] : 0.f;
    Xb[idx] = f2bf(v);
}

// W (K x N f32, row-major) -> Wt (N x Kpad bf16), transposed + zero-padded.
__global__ __launch_bounds__(256) void k_transpose_w(const float* __restrict__ W,
                                                     __bf16* __restrict__ Wt,
                                                     int K, int N, int Kpad) {
    int idx = blockIdx.x * 256 + threadIdx.x;              // over N*Kpad
    int n = idx / Kpad, k = idx - n * Kpad;
    float v = (k < K) ? W[(size_t)k * N + n] : 0.f;
    Wt[idx] = f2bf(v);
}

// Mt[j][i] = (j>i) ? sum_k nfk[i,f2f[j],k]*nfk[j,f2f[i],k] : 0   (transposed M)
__global__ __launch_bounds__(256) void k_build_mt(const float* __restrict__ nfk,
                                                  const int* __restrict__ f2f,
                                                  __bf16* __restrict__ Mt) {
    int idx = blockIdx.x * 256 + threadIdx.x;              // over KPAD*KPAD
    int j = idx >> 10, i = idx & 1023;
    float s = 0.f;
    if (i < FSZ && j < FSZ && j > i) {
        int fi = f2f[i], fj = f2f[j];
        const float* pi = nfk + ((size_t)i * NFIELDS + fj) * KFF;
        const float* pj = nfk + ((size_t)j * NFIELDS + fi) * KFF;
        #pragma unroll
        for (int k = 0; k < KFF; ++k) s += pi[k] * pj[k];
    }
    Mt[idx] = f2bf(s);
}

// ------------------------------- WMMA GEMM ---------------------------------
// C[M,N] = op(A[M,K] @ Bt[N,K]^T + bias), block tile 128x128, K-step 32.
// 8 waves: wave w -> (wm = w&3)*32 rows, (wn = w>>2)*64 cols; 2x4 WMMA tiles.
// TDM stages tiles into double-buffered LDS: wave0 moves A, wave1 moves B.
__global__ __launch_bounds__(256)
void k_gemm_bf16(const __bf16* __restrict__ A, const __bf16* __restrict__ Bt,
                 const float* __restrict__ bias,
                 __bf16* __restrict__ Cb, float* __restrict__ Cf,
                 int Ksteps, int lda, int ldb, int ldc, int relu)
{
    __shared__ __align__(16) __bf16 As[2][128 * 32];
    __shared__ __align__(16) __bf16 Bs[2][128 * 32];

    const int t    = threadIdx.x;
    const int w    = t >> 5;
    const int lane = t & 31;
    const int lr   = lane & 15;
    const int half = lane >> 4;
    const int wm   = w & 3;
    const int wn   = w >> 2;
    const int bm   = blockIdx.y * 128;
    const int bn   = blockIdx.x * 128;

    // LDS byte addresses (generic-pointer low 32 bits == LDS offset)
    const uint32_t ldsA0 = (uint32_t)(uintptr_t)(void*)&As[0][0];
    const uint32_t ldsA1 = (uint32_t)(uintptr_t)(void*)&As[1][0];
    const uint32_t ldsB0 = (uint32_t)(uintptr_t)(void*)&Bs[0][0];
    const uint32_t ldsB1 = (uint32_t)(uintptr_t)(void*)&Bs[1][0];

    const v8f vzero = {0.f, 0.f, 0.f, 0.f, 0.f, 0.f, 0.f, 0.f};
    v8f acc[2][4];
    #pragma unroll
    for (int mi = 0; mi < 2; ++mi)
        #pragma unroll
        for (int ni = 0; ni < 4; ++ni) acc[mi][ni] = vzero;

    union FragU { uint4 u[2]; v16bf v; };

    // prologue: DMA first tile pair into buffer 0
    if (w == 0) tdm_load_tile(A,  (size_t)bm * lda, lda, ldsA0);
    if (w == 1) tdm_load_tile(Bt, (size_t)bn * ldb, ldb, ldsB0);

    for (int ks = 0; ks < Ksteps; ++ks) {
        const int buf = ks & 1;
        if (ks + 1 < Ksteps) {
            // DMA next tile pair into the other buffer (its readers finished
            // at the trailing barrier of iteration ks-1)
            if (w == 0) tdm_load_tile(A,  (size_t)bm * lda + (size_t)(ks + 1) * 32,
                                      lda, buf ? ldsA0 : ldsA1);
            if (w == 1) tdm_load_tile(Bt, (size_t)bn * ldb + (size_t)(ks + 1) * 32,
                                      ldb, buf ? ldsB0 : ldsB1);
            if (w < 2) __builtin_amdgcn_s_wait_tensorcnt((short)1); // current done
        } else {
            if (w < 2) __builtin_amdgcn_s_wait_tensorcnt((short)0);
        }
        __syncthreads();   // release: current buffer is valid for all waves

        FragU af[2], bfr[4];
        // A 16x32 bf16 layout: lane-half h: VGPR0-3 <-> K = h*8..h*8+7,
        //                                   VGPR4-7 <-> K = 16+h*8..16+h*8+7
        #pragma unroll
        for (int mi = 0; mi < 2; ++mi) {
            const __bf16* p = &As[buf][(wm * 32 + mi * 16 + lr) * 32 + half * 8];
            af[mi].u[0] = *(const uint4*)p;
            af[mi].u[1] = *(const uint4*)(p + 16);
        }
        // B 32x16 bf16 layout: lane-half h: VGPR0-3 <-> K = h*8..h*8+7,
        //                                   VGPR4-7 <-> K = 16+h*8..16+h*8+7
        #pragma unroll
        for (int ni = 0; ni < 4; ++ni) {
            const __bf16* p = &Bs[buf][(wn * 64 + ni * 16 + lr) * 32 + half * 8];
            bfr[ni].u[0] = *(const uint4*)p;
            bfr[ni].u[1] = *(const uint4*)(p + 16);
        }
        #pragma unroll
        for (int mi = 0; mi < 2; ++mi)
            #pragma unroll
            for (int ni = 0; ni < 4; ++ni)
                acc[mi][ni] = __builtin_amdgcn_wmma_f32_16x16x32_bf16(
                    false, af[mi].v, false, bfr[ni].v,
                    (short)0, acc[mi][ni], false, false);

        __syncthreads();   // all waves done reading buf before TDM overwrites it
    }

    // epilogue: D 16x16 f32 layout: VGPR v = row (half*8+v), lane lr = col
    #pragma unroll
    for (int ni = 0; ni < 4; ++ni) {
        int col = bn + wn * 64 + ni * 16 + lr;
        float bv = bias ? bias[col] : 0.f;
        #pragma unroll
        for (int mi = 0; mi < 2; ++mi) {
            #pragma unroll
            for (int v = 0; v < 8; ++v) {
                int row = bm + wm * 32 + mi * 16 + half * 8 + v;
                float val = acc[mi][ni][v] + bv;
                if (relu && val < 0.f) val = 0.f;
                if (Cb) Cb[(size_t)row * ldc + col] = f2bf(val);
                if (Cf) Cf[(size_t)row * ldc + col] = val;
            }
        }
    }
}

// ------------------------------- finalize ----------------------------------
// out[b] = X[b]·w1 + b0 + sum_f XM[b,f]*X[b,f] + H2[b]·outW + outB
__global__ __launch_bounds__(256)
void k_finalize(const float* __restrict__ X, const float* __restrict__ w1,
                const float* __restrict__ bscalar, const float* __restrict__ XM,
                const __bf16* __restrict__ H2, const float* __restrict__ outW,
                const float* __restrict__ outB, float* __restrict__ out)
{
    int b = blockIdx.x, t = threadIdx.x;
    float s = 0.f;
    for (int f = t; f < FSZ; f += 256) {
        float x = X[(size_t)b * FSZ + f];
        s += x * w1[f];                           // linear partial
        s += XM[(size_t)b * KPAD + f] * x;        // interaction partial
    }
    if (t < L2) s += bf2f(H2[(size_t)b * L2 + t]) * outW[t];   // deep partial
    __shared__ float red[256];
    red[t] = s;
    __syncthreads();
    for (int off = 128; off > 0; off >>= 1) {
        if (t < off) red[t] += red[t + off];
        __syncthreads();
    }
    if (t == 0) out[b] = red[0] + bscalar[0] + outB[0];
}

// ------------------------------ launcher -----------------------------------

extern "C" void kernel_launch(void* const* d_in, const int* in_sizes, int n_in,
                              void* d_out, int out_size, void* d_ws, size_t ws_size,
                              hipStream_t stream) {
    (void)in_sizes; (void)n_in; (void)out_size; (void)ws_size;

    const float* X      = (const float*)d_in[0];
    const float* w1     = (const float*)d_in[1];
    const float* bsc    = (const float*)d_in[2];
    const float* nfk    = (const float*)d_in[3];
    const int*   f2f    = (const int*)  d_in[4];
    const float* deepW0 = (const float*)d_in[5];
    const float* deepB0 = (const float*)d_in[6];
    const float* deepW1 = (const float*)d_in[7];
    const float* deepB1 = (const float*)d_in[8];
    const float* deepW2 = (const float*)d_in[9];
    const float* deepB2 = (const float*)d_in[10];
    const float* outW   = (const float*)d_in[11];
    const float* outB   = (const float*)d_in[12];
    float* out = (float*)d_out;

    // workspace partition (256B aligned)
    char*  base = (char*)d_ws;
    size_t off  = 0;
    auto carve = [&](size_t bytes) -> char* {
        char* p = base + off;
        off = (off + bytes + 255) & ~(size_t)255;
        return p;
    };
    __bf16* Xb  = (__bf16*)carve((size_t)BATCH * KPAD * 2);
    __bf16* W0t = (__bf16*)carve((size_t)L0 * KPAD * 2);
    __bf16* W1t = (__bf16*)carve((size_t)L1 * KPAD * 2);
    __bf16* W2t = (__bf16*)carve((size_t)L2 * 512  * 2);
    __bf16* Mt  = (__bf16*)carve((size_t)KPAD * KPAD * 2);
    __bf16* H0  = (__bf16*)carve((size_t)BATCH * L0 * 2);
    __bf16* H1  = (__bf16*)carve((size_t)BATCH * L1 * 2);
    __bf16* H2  = (__bf16*)carve((size_t)BATCH * L2 * 2);
    float*  XM  = (float*) carve((size_t)BATCH * KPAD * 4);

    // 1) conversions (bf16, padded, weights transposed to N x Kpad)
    k_pad_x<<<(BATCH * KPAD) / 256, 256, 0, stream>>>(X, Xb);
    k_transpose_w<<<(L0 * KPAD) / 256, 256, 0, stream>>>(deepW0, W0t, FSZ, L0, KPAD);
    k_transpose_w<<<(L1 * KPAD) / 256, 256, 0, stream>>>(deepW1, W1t, L0, L1, KPAD);
    k_transpose_w<<<(L2 * 512 ) / 256, 256, 0, stream>>>(deepW2, W2t, L1, L2, 512);
    k_build_mt<<<(KPAD * KPAD) / 256, 256, 0, stream>>>(nfk, f2f, Mt);

    // 2) WMMA GEMMs: grid (N/128, M/128), Ksteps = K/32
    dim3 blk(256);
    // XM = Xb @ M        (f32 out, no bias/relu)
    k_gemm_bf16<<<dim3(KPAD / 128, BATCH / 128), blk, 0, stream>>>(
        Xb, Mt, nullptr, nullptr, XM, KPAD / 32, KPAD, KPAD, KPAD, 0);
    // H0 = relu(Xb @ W0 + b0)
    k_gemm_bf16<<<dim3(L0 / 128, BATCH / 128), blk, 0, stream>>>(
        Xb, W0t, deepB0, H0, nullptr, KPAD / 32, KPAD, KPAD, L0, 1);
    // H1 = relu(H0 @ W1 + b1)
    k_gemm_bf16<<<dim3(L1 / 128, BATCH / 128), blk, 0, stream>>>(
        H0, W1t, deepB1, H1, nullptr, L0 / 32, L0, KPAD, L1, 1);
    // H2 = relu(H1 @ W2 + b2)
    k_gemm_bf16<<<dim3(L2 / 128, BATCH / 128), blk, 0, stream>>>(
        H1, W2t, deepB2, H2, nullptr, L1 / 32, L1, 512, L2, 1);

    // 3) row-wise reduction: linear + interaction + deep head
    k_finalize<<<BATCH, 256, 0, stream>>>(X, w1, bsc, XM, H2, outW, outB, out);
}
